// WaveRNNCell_481036337285
// MI455X (gfx1250) — matrix operand order repaired and checked by reading, not compiled, not run on hardware
//
#include <hip/hip_runtime.h>
#include <hip/hip_bf16.h>

// ---------------------------------------------------------------------------
// WaveRNN cell on MI455X (gfx1250): bf16 WMMA GEMMs, async global->LDS
// staging (GLOBAL_LOAD_ASYNC_TO_LDS_B128), double-buffered K pipeline.
// ---------------------------------------------------------------------------

typedef __attribute__((ext_vector_type(16))) __bf16 v16bf;
typedef __attribute__((ext_vector_type(8)))  float  v8f;
typedef __attribute__((ext_vector_type(4)))  int    v4i;

static constexpr int Bsz  = 16384;
static constexpr int Hh   = 1024;
static constexpr int FEAT = 128;
static constexpr int AUX  = 64;
static constexpr int FCn  = 1024;
static constexpr int Cc   = 256;
static constexpr int KIN  = 256;            // 194 (FEAT+AUX+2) padded to mult of 64
static constexpr int KFC  = 576;            // H/2 + AUX = 512 + 64 = 9*64

#if defined(__gfx1250__) && __has_builtin(__builtin_amdgcn_global_load_async_to_lds_b128)
#define USE_ASYNC_LDS 1
#else
#define USE_ASYNC_LDS 0
#endif

#if USE_ASYNC_LDS
typedef __attribute__((address_space(1))) v4i gv4i;   // global 16B chunk
typedef __attribute__((address_space(3))) v4i lv4i;   // LDS 16B chunk
#endif

__device__ __forceinline__ unsigned short f2bf(float f) {
  unsigned u = __float_as_uint(f);
  u += 0x7fffu + ((u >> 16) & 1u);          // round-to-nearest-even
  return (unsigned short)(u >> 16);
}

// 16-byte global -> LDS copy (async on CDNA5, sync fallback elsewhere)
__device__ __forceinline__ void cp16_g2l(const unsigned short* g, unsigned short* l) {
#if USE_ASYNC_LDS
  __builtin_amdgcn_global_load_async_to_lds_b128((gv4i*)g, (lv4i*)l, 0, 0);
#else
  *(uint4*)l = *(const uint4*)g;
#endif
}

__device__ __forceinline__ void drain_async() {
#if USE_ASYNC_LDS
#if __has_builtin(__builtin_amdgcn_s_wait_asynccnt)
  __builtin_amdgcn_s_wait_asynccnt(0);
#else
  asm volatile("s_wait_asynccnt 0x0" ::: "memory");
#endif
#endif
}

// ------------------------- conversion kernels ------------------------------

// build bf16 [B, KIN] = [feat(128) | aux1(64) | x(2) | zeros]
__global__ void k_concat_inp(const float* __restrict__ feat,
                             const float* __restrict__ aux1,
                             const float* __restrict__ x,
                             unsigned short* __restrict__ out) {
  long i = (long)blockIdx.x * 256 + threadIdx.x;
  if (i >= (long)Bsz * KIN) return;
  long b = i / KIN; int c = (int)(i % KIN);
  float v = 0.f;
  if (c < FEAT)                v = feat[b * FEAT + c];
  else if (c < FEAT + AUX)     v = aux1[b * AUX + (c - FEAT)];
  else if (c < FEAT + AUX + 2) v = x[b * 2 + (c - FEAT - AUX)];
  out[i] = f2bf(v);
}

__global__ void k_cvt(const float* __restrict__ src,
                      unsigned short* __restrict__ dst, long n) {
  long i = (long)blockIdx.x * 256 + threadIdx.x;
  if (i < n) dst[i] = f2bf(src[i]);
}

// fp32 [rows, sc] -> bf16 [rows, dc] with zero pad (for w_ih 194 -> 256)
__global__ void k_cvt_pad(const float* __restrict__ src,
                          unsigned short* __restrict__ dst,
                          int rows, int sc, int dc) {
  long i = (long)blockIdx.x * 256 + threadIdx.x;
  if (i >= (long)rows * dc) return;
  long r = i / dc; int c = (int)(i % dc);
  dst[i] = (c < sc) ? f2bf(src[r * sc + c]) : (unsigned short)0;
}

// ------------------------------ WMMA GEMM ----------------------------------
// C[M,N] = A[M,K](bf16,row-major) x W[N,K](bf16,row-major)^T
// mode 0: Cf = acc                      (fp32, no bias)
// mode 1: Cbf = bf16(relu(acc + bias))
// mode 2: Cf = acc + bias               (fp32)
// Requires M%128==0, N%128==0, K%64==0.
__global__ __launch_bounds__(256)
void gemm_bf16(const unsigned short* __restrict__ A,
               const unsigned short* __restrict__ W,
               const float* __restrict__ bias,
               float* __restrict__ Cf,
               unsigned short* __restrict__ Cbf,
               int M, int N, int K, int ldc, int mode) {
  constexpr int BK  = 64;                   // K tile
  constexpr int LDT = 72;                   // padded LDS row stride (elems)
  __shared__ unsigned short sA[2][128 * LDT];
  __shared__ unsigned short sW[2][128 * LDT];

  const int  tid  = threadIdx.x;
  const int  lane = tid & 31;
  const int  wid  = tid >> 5;               // 8 waves
  const int  wm   = wid >> 2;               // 0..1  -> 64 rows each
  const int  wn   = wid & 3;                // 0..3  -> 32 cols each
  const long m0   = (long)blockIdx.y * 128;
  const long n0   = (long)blockIdx.x * 128;

  v8f acc[4][2];
  const v8f vzero = {0.f, 0.f, 0.f, 0.f, 0.f, 0.f, 0.f, 0.f};
#pragma unroll
  for (int mi = 0; mi < 4; ++mi)
#pragma unroll
    for (int ni = 0; ni < 2; ++ni) acc[mi][ni] = vzero;

  union Frag { v16bf v; uint4 q[2]; };

  const int am = lane & 15;                 // M (or N) index within 16
  const int ak = (lane >> 4) * 8;           // A frag K chunk base: 0 or 8
  const int wk = (lane >> 4) * 8;           // B frag K chunk base: 0 or 8

  // stage one 128xBK tile of A and W into LDS buffer `buf`
  auto stage = [&](int buf, int k0) {
#pragma unroll
    for (int i = 0; i < 4; ++i) {
      int c   = tid + i * 256;              // 0..1023 chunks of 8 elems
      int row = c >> 3;                     // 0..127
      int col = (c & 7) * 8;                // 0..56
      cp16_g2l(&A[(m0 + row) * (long)K + k0 + col], &sA[buf][row * LDT + col]);
      cp16_g2l(&W[(n0 + row) * (long)K + k0 + col], &sW[buf][row * LDT + col]);
    }
  };

  const int nk = K / BK;
  stage(0, 0);

  for (int t = 0; t < nk; ++t) {
    drain_async();                          // my async copies into buf t&1 done
    __syncthreads();                        // everyone's copies done + prev compute done
    if (t + 1 < nk) stage((t + 1) & 1, (t + 1) * BK);

    const unsigned short* aB = &sA[t & 1][0];
    const unsigned short* wB = &sW[t & 1][0];
#pragma unroll
    for (int kk = 0; kk < BK; kk += 32) {
      Frag af[4];
#pragma unroll
      for (int mi = 0; mi < 4; ++mi) {
        const unsigned short* p = &aB[(wm * 64 + mi * 16 + am) * LDT + kk];
        af[mi].q[0] = *(const uint4*)&p[ak];
        af[mi].q[1] = *(const uint4*)&p[ak + 16];
      }
      Frag wf[2];
#pragma unroll
      for (int ni = 0; ni < 2; ++ni) {
        const unsigned short* p = &wB[(wn * 32 + ni * 16 + am) * LDT + kk + wk];
        wf[ni].q[0] = *(const uint4*)&p[0];
        wf[ni].q[1] = *(const uint4*)&p[16];
      }
#pragma unroll
      for (int mi = 0; mi < 4; ++mi)
#pragma unroll
        for (int ni = 0; ni < 2; ++ni)
          acc[mi][ni] = __builtin_amdgcn_wmma_f32_16x16x32_bf16(
              false, af[mi].v, false, wf[ni].v, (short)0, acc[mi][ni],
              false, false);
    }
    __syncthreads();                        // done reading buf t&1
  }

  // epilogue: VGPR r -> row r + 8*(lane>>4); col = lane&15
  const int cn    = lane & 15;
  const int rbase = (lane >> 4) * 8;
#pragma unroll
  for (int mi = 0; mi < 4; ++mi)
#pragma unroll
    for (int ni = 0; ni < 2; ++ni) {
      long col = n0 + wn * 32 + ni * 16 + cn;
      float bv = (mode != 0) ? bias[col] : 0.f;
#pragma unroll
      for (int r = 0; r < 8; ++r) {
        long row = m0 + wm * 64 + mi * 16 + rbase + r;
        float v  = acc[mi][ni][r] + bv;
        if (mode == 1) {
          v = v > 0.f ? v : 0.f;
          Cbf[row * ldc + col] = f2bf(v);
        } else {
          Cf[row * ldc + col] = v;
        }
      }
    }
}

// --------------------------- GRU elementwise -------------------------------
__global__ void k_gru(const float* __restrict__ gi, const float* __restrict__ gh,
                      const float* __restrict__ b_ih, const float* __restrict__ b_hh,
                      const float* __restrict__ h, float* __restrict__ h1,
                      unsigned short* __restrict__ fcin_c,
                      unsigned short* __restrict__ fcin_f) {
  long i = (long)blockIdx.x * 256 + threadIdx.x;
  if (i >= (long)Bsz * Hh) return;
  long b = i / Hh; int j = (int)(i % Hh);
  const float* gib = gi + b * 3 * Hh;
  const float* ghb = gh + b * 3 * Hh;
  float ir = gib[j]          + b_ih[j];
  float iz = gib[Hh + j]     + b_ih[Hh + j];
  float in_= gib[2*Hh + j]   + b_ih[2*Hh + j];
  float hr = ghb[j]          + b_hh[j];
  float hz = ghb[Hh + j]     + b_hh[Hh + j];
  float hn = ghb[2*Hh + j]   + b_hh[2*Hh + j];
  float r = 1.f / (1.f + __expf(-(ir + hr)));
  float z = 1.f / (1.f + __expf(-(iz + hz)));
  float n = tanhf(in_ + r * hn);
  float hv = (1.f - z) * n + z * h[i];
  h1[i] = hv;
  if (j < Hh / 2) fcin_c[b * KFC + j]            = f2bf(hv);
  else            fcin_f[b * KFC + (j - Hh / 2)] = f2bf(hv);
}

__global__ void k_aux(const float* __restrict__ aux2, const float* __restrict__ aux3,
                      unsigned short* __restrict__ fcin_c,
                      unsigned short* __restrict__ fcin_f) {
  long i = (long)blockIdx.x * 256 + threadIdx.x;
  if (i >= (long)Bsz * AUX) return;
  long b = i / AUX; int t = (int)(i % AUX);
  fcin_c[b * KFC + Hh / 2 + t] = f2bf(aux2[i]);
  fcin_f[b * KFC + Hh / 2 + t] = f2bf(aux3[i]);
}

// ------------------------------- launcher ----------------------------------
extern "C" void kernel_launch(void* const* d_in, const int* in_sizes, int n_in,
                              void* d_out, int out_size, void* d_ws, size_t ws_size,
                              hipStream_t stream) {
  const float* x     = (const float*)d_in[0];
  const float* feat  = (const float*)d_in[1];
  const float* aux1  = (const float*)d_in[2];
  const float* aux2  = (const float*)d_in[3];
  const float* aux3  = (const float*)d_in[4];
  const float* h     = (const float*)d_in[5];
  const float* w_ih  = (const float*)d_in[6];
  const float* w_hh  = (const float*)d_in[7];
  const float* b_ih  = (const float*)d_in[8];
  const float* b_hh  = (const float*)d_in[9];
  const float* fc1_w = (const float*)d_in[10];
  const float* fc1_b = (const float*)d_in[11];
  const float* fc2_w = (const float*)d_in[12];
  const float* fc2_b = (const float*)d_in[13];
  const float* fc3_w = (const float*)d_in[14];
  const float* fc3_b = (const float*)d_in[15];
  const float* fc4_w = (const float*)d_in[16];
  const float* fc4_b = (const float*)d_in[17];

  float* o_c = (float*)d_out;                       // [B, 256]
  float* o_f = o_c + (size_t)Bsz * Cc;              // [B, 256]
  float* h1  = o_c + (size_t)Bsz * Cc * 2;          // [B, 1024]

  // workspace carve-up
  char*  base = (char*)d_ws;
  size_t off  = 0;
  auto take = [&](size_t bytes) -> void* {
    void* r = base + off;
    off = (off + bytes + 255) & ~(size_t)255;
    return r;
  };
  unsigned short* inp_bf   = (unsigned short*)take((size_t)Bsz * KIN * 2);
  unsigned short* h_bf     = (unsigned short*)take((size_t)Bsz * Hh * 2);
  unsigned short* w_ih_bf  = (unsigned short*)take((size_t)3 * Hh * KIN * 2);
  unsigned short* w_hh_bf  = (unsigned short*)take((size_t)3 * Hh * Hh * 2);
  unsigned short* fc1_w_bf = (unsigned short*)take((size_t)FCn * KFC * 2);
  unsigned short* fc2_w_bf = (unsigned short*)take((size_t)Cc * FCn * 2);
  unsigned short* fc3_w_bf = (unsigned short*)take((size_t)FCn * KFC * 2);
  unsigned short* fc4_w_bf = (unsigned short*)take((size_t)Cc * FCn * 2);
  float*          gi_ws    = (float*)take((size_t)Bsz * 3 * Hh * 4);
  float*          gh_ws    = (float*)take((size_t)Bsz * 3 * Hh * 4);
  unsigned short* fcin_c   = (unsigned short*)take((size_t)Bsz * KFC * 2);
  unsigned short* fcin_f   = (unsigned short*)take((size_t)Bsz * KFC * 2);
  unsigned short* hid_c    = (unsigned short*)take((size_t)Bsz * FCn * 2);
  unsigned short* hid_f    = (unsigned short*)take((size_t)Bsz * FCn * 2);

  const dim3 blk(256);
  auto blocks1d = [](long n) { return dim3((unsigned)((n + 255) / 256)); };

  // 1) fp32 -> bf16 staging
  k_concat_inp<<<blocks1d((long)Bsz * KIN), blk, 0, stream>>>(feat, aux1, x, inp_bf);
  k_cvt<<<blocks1d((long)Bsz * Hh), blk, 0, stream>>>(h, h_bf, (long)Bsz * Hh);
  k_cvt_pad<<<blocks1d((long)3 * Hh * KIN), blk, 0, stream>>>(w_ih, w_ih_bf, 3 * Hh, FEAT + AUX + 2, KIN);
  k_cvt<<<blocks1d((long)3 * Hh * Hh), blk, 0, stream>>>(w_hh, w_hh_bf, (long)3 * Hh * Hh);
  k_cvt<<<blocks1d((long)FCn * KFC), blk, 0, stream>>>(fc1_w, fc1_w_bf, (long)FCn * KFC);
  k_cvt<<<blocks1d((long)Cc * FCn), blk, 0, stream>>>(fc2_w, fc2_w_bf, (long)Cc * FCn);
  k_cvt<<<blocks1d((long)FCn * KFC), blk, 0, stream>>>(fc3_w, fc3_w_bf, (long)FCn * KFC);
  k_cvt<<<blocks1d((long)Cc * FCn), blk, 0, stream>>>(fc4_w, fc4_w_bf, (long)Cc * FCn);

  // 2) GRU gate GEMMs (fp32 accum out, biases folded in elementwise stage)
  gemm_bf16<<<dim3(3 * Hh / 128, Bsz / 128), blk, 0, stream>>>(
      inp_bf, w_ih_bf, nullptr, gi_ws, nullptr, Bsz, 3 * Hh, KIN, 3 * Hh, 0);
  gemm_bf16<<<dim3(3 * Hh / 128, Bsz / 128), blk, 0, stream>>>(
      h_bf, w_hh_bf, nullptr, gh_ws, nullptr, Bsz, 3 * Hh, Hh, 3 * Hh, 0);

  // 3) GRU elementwise -> h1 (fp32 out) + bf16 fc-head inputs
  k_gru<<<blocks1d((long)Bsz * Hh), blk, 0, stream>>>(
      gi_ws, gh_ws, b_ih, b_hh, h, h1, fcin_c, fcin_f);
  k_aux<<<blocks1d((long)Bsz * AUX), blk, 0, stream>>>(aux2, aux3, fcin_c, fcin_f);

  // 4) heads: fc1/fc3 (bias+relu -> bf16), fc2/fc4 (bias -> fp32 out)
  gemm_bf16<<<dim3(FCn / 128, Bsz / 128), blk, 0, stream>>>(
      fcin_c, fc1_w_bf, fc1_b, nullptr, hid_c, Bsz, FCn, KFC, FCn, 1);
  gemm_bf16<<<dim3(FCn / 128, Bsz / 128), blk, 0, stream>>>(
      fcin_f, fc3_w_bf, fc3_b, nullptr, hid_f, Bsz, FCn, KFC, FCn, 1);
  gemm_bf16<<<dim3(Cc / 128, Bsz / 128), blk, 0, stream>>>(
      hid_c, fc2_w_bf, fc2_b, o_c, nullptr, Bsz, Cc, FCn, Cc, 2);
  gemm_bf16<<<dim3(Cc / 128, Bsz / 128), blk, 0, stream>>>(
      hid_f, fc4_w_bf, fc4_b, o_f, nullptr, Bsz, Cc, FCn, Cc, 2);

  (void)in_sizes; (void)n_in; (void)out_size; (void)ws_size;
}
